// EncodeBlock_13735305412877
// MI455X (gfx1250) — compile-verified, matrix operand order checked by reading, not run on hardware
//
#include <hip/hip_runtime.h>
#include <math.h>

typedef __attribute__((ext_vector_type(16))) __bf16 v16bf;
typedef __attribute__((ext_vector_type(8)))  float  v8f;

// ---------------------------------------------------------------------------
// WMMA helper: D = A(16x32 bf16) * B(32x16 bf16) + C(16x16 f32)
// ---------------------------------------------------------------------------
__device__ inline v8f wmma_bf16(v16bf a, v16bf b, v8f c) {
  return __builtin_amdgcn_wmma_f32_16x16x32_bf16(
      /*neg_a=*/false, a, /*neg_b=*/false, b,
      /*c_mod=*/(short)0, c, /*reuse_a=*/false, /*reuse_b=*/false);
}

// Fragment element pattern (16-bit A/B, 16x32): per lane, element e maps to
// K = kh + e (e<8) or K = 16 + kh + (e-8) (e>=8), with kh = 8*(lane>=16).
__device__ inline v16bf load_frag_bf16(const __bf16* base, int kh) {
  v16bf r;
#pragma unroll
  for (int e = 0; e < 8; ++e) { r[e] = base[kh + e]; r[e + 8] = base[16 + kh + e]; }
  return r;
}

__device__ inline v16bf cvt_frag_f32(const float* base, int kh) {
  v16bf r;
#pragma unroll
  for (int e = 0; e < 8; ++e) { r[e] = (__bf16)base[kh + e]; r[e + 8] = (__bf16)base[16 + kh + e]; }
  return r;
}

// Low 32 bits of a flat shared-aperture address = LDS byte offset.
__device__ inline unsigned int lds_addr_u32(const void* p) {
  return (unsigned int)(size_t)p;
}

// CDNA5 async copy: 16B global -> LDS per lane, tracked by ASYNCcnt.
__device__ inline void async_load_b128(unsigned int lds_byte_addr, const void* gptr) {
  unsigned long long ga = (unsigned long long)(size_t)gptr;
  asm volatile("global_load_async_to_lds_b128 %0, %1, off"
               :
               : "v"(lds_byte_addr), "v"(ga)
               : "memory");
}
__device__ inline void wait_asynccnt0() {
  asm volatile("s_wait_asynccnt 0x0" ::: "memory");
}

// Row reductions: a C-matrix row lives in 16 lanes of one half-wave.
__device__ inline float rowmax16(float v) {
  v = fmaxf(v, __shfl_xor(v, 1, 32));
  v = fmaxf(v, __shfl_xor(v, 2, 32));
  v = fmaxf(v, __shfl_xor(v, 4, 32));
  v = fmaxf(v, __shfl_xor(v, 8, 32));
  return v;
}
__device__ inline float rowsum16(float v) {
  v += __shfl_xor(v, 1, 32);
  v += __shfl_xor(v, 2, 32);
  v += __shfl_xor(v, 4, 32);
  v += __shfl_xor(v, 8, 32);
  return v;
}

// ---------------------------------------------------------------------------
// fp32 -> bf16 elementwise
// ---------------------------------------------------------------------------
__global__ __launch_bounds__(256) void cvt_kernel(const float* __restrict__ in,
                                                  __bf16* __restrict__ out, int n) {
  int i = blockIdx.x * 256 + threadIdx.x;
  if (i < n) out[i] = (__bf16)in[i];
}

// W[K x N] row-major fp32  ->  Wt[N x K] row-major bf16
__global__ __launch_bounds__(256) void transpose_cvt_kernel(const float* __restrict__ in,
                                                            __bf16* __restrict__ out,
                                                            int K, int N) {
  int i = blockIdx.x * 256 + threadIdx.x;
  if (i < K * N) {
    int k = i / N, n = i % N;
    out[(size_t)n * K + k] = (__bf16)in[i];
  }
}

// ---------------------------------------------------------------------------
// GEMM: out[M x N] = A_bf16[M x K] @ Bt_bf16[N x K]^T + bias, optional relu.
// Wave = 16x64 output strip; 8 waves/block cover 128 rows x 64 cols.
// The shared 64x32 B tile is staged into LDS with CDNA5 async-to-LDS copies
// (256 lanes x 16B = 4KB per k-step), then read as ds_load_b128 fragments.
// grid = (N/64, M/128).
// ---------------------------------------------------------------------------
__global__ __launch_bounds__(256) void gemm_kernel(const __bf16* __restrict__ A,
                                                   const __bf16* __restrict__ Bt,
                                                   const float* __restrict__ bias,
                                                   float* outF, __bf16* outB,
                                                   int M, int N, int K, int relu) {
  __shared__ __bf16 btile[64 * 32];  // [col_local][k_local], 4KB

  const int tid = threadIdx.x;
  const int lane = tid & 31;
  const int wave = tid >> 5;
  const int lanelo = lane & 15;
  const int half = lane >> 4;
  const int kh = half * 8;

  const int m0 = blockIdx.y * 128 + wave * 16;
  const int n0 = blockIdx.x * 64;
  const int row = m0 + lanelo;

  // Staging assignment: one 16B chunk per thread (64 cols x 4 chunks).
  const int scol = tid >> 2;           // 0..63
  const int skoff = (tid & 3) * 8;     // 0,8,16,24
  const unsigned int sdst = lds_addr_u32(btile) + (unsigned int)((scol * 32 + skoff) * 2);

  v8f zero = {};
  v8f acc[4] = {zero, zero, zero, zero};

  for (int k0 = 0; k0 < K; k0 += 32) {
    // Async-stage B tile: btile[c][k] = Bt[n0+c][k0+k]
    async_load_b128(sdst, Bt + (size_t)(n0 + scol) * K + k0 + skoff);
    // Prefetch next A tile while the copy is in flight.
    if (k0 + 32 < K) __builtin_prefetch(A + (size_t)row * K + k0 + 32, 0, 1);

    v16bf va = load_frag_bf16(A + (size_t)row * K + k0, kh);

    wait_asynccnt0();
    __syncthreads();

#pragma unroll
    for (int t = 0; t < 4; ++t) {
      v16bf vb = load_frag_bf16(btile + (t * 16 + lanelo) * 32, kh);
      acc[t] = wmma_bf16(va, vb, acc[t]);
    }
    __syncthreads();  // protect btile before next k-step's async stores
  }

#pragma unroll
  for (int t = 0; t < 4; ++t) {
    int col = n0 + t * 16 + lanelo;
    float bv = bias[col];
#pragma unroll
    for (int r = 0; r < 8; ++r) {
      int mrow = m0 + r + 8 * half;
      float v = acc[t][r] + bv;
      if (relu) v = fmaxf(v, 0.0f);
      size_t off = (size_t)mrow * N + col;
      if (outF) outF[off] = v;
      if (outB) outB[off] = (__bf16)v;
    }
  }
}

// ---------------------------------------------------------------------------
// Attention (q = k = v = Q, per the reference's shared projection).
// Block = 4 waves x 16 query rows of one (b,h). Per 32-key step the block
// cooperatively stages the V tile into LDS transposed+converted to bf16
// ([d][key]) so every V B-fragment is two contiguous ds_load_b128.
// S = Q Q^T / 8 via 4 WMMAs; P relayout via per-wave LDS; O += P V via 4.
// grid = (1024/64, b*h = 64). Output Z in concat layout as bf16.
// ---------------------------------------------------------------------------
__global__ __launch_bounds__(128) void attn_kernel(const float* __restrict__ Q,
                                                   __bf16* __restrict__ Zbf) {
  __shared__ __bf16 vtile[64 * 32];      // [d][key], 4KB
  __shared__ __bf16 pLds[4 * 16 * 32];   // per-wave P, 4KB

  const int tid = threadIdx.x;
  const int lane = tid & 31;
  const int wave = tid >> 5;
  const int lanelo = lane & 15;
  const int half = lane >> 4;
  const int kh = half * 8;

  const int b = blockIdx.y >> 4;
  const int h = blockIdx.y & 15;
  const int q0 = blockIdx.x * 64 + wave * 16;
  const float* Qh = Q + (size_t)b * 1024 * 1024 + h * 64;  // (s,d) at s*1024 + d
  __bf16* pbase = pLds + wave * 16 * 32;

  // Staging assignment: thread reads 16 consecutive floats of one key row.
  const int skey = tid >> 2;           // 0..31
  const int sdoff = (tid & 3) * 16;    // 0,16,32,48

  // Preload Q A-fragments for d-steps 0 and 32 (d_k = 64).
  v16bf aq0, aq1;
  {
    const float* rp = Qh + (size_t)(q0 + lanelo) * 1024;
    aq0 = cvt_frag_f32(rp, kh);
    aq1 = cvt_frag_f32(rp + 32, kh);
  }

  float m[8], l[8];
  v8f zero = {};
  v8f o[4] = {zero, zero, zero, zero};
#pragma unroll
  for (int r = 0; r < 8; ++r) { m[r] = -1.0e30f; l[r] = 0.0f; }

  const float inv_scale = 0.125f;  // 1/sqrt(64)

  for (int j0 = 0; j0 < 1024; j0 += 32) {
    // Cooperative V-tile stage: vtile[d][key] = bf16(Qh[j0+key][d])
    {
      const float* src = Qh + (size_t)(j0 + skey) * 1024 + sdoff;
#pragma unroll
      for (int i = 0; i < 16; ++i)
        vtile[(sdoff + i) * 32 + skey] = (__bf16)src[i];
    }
    __syncthreads();

    // S tile: 16 queries x 32 keys (two 16x16 C fragments)
    v8f s0 = zero, s1 = zero;
    {
      const float* kp = Qh + (size_t)(j0 + lanelo) * 1024;
      s0 = wmma_bf16(aq0, cvt_frag_f32(kp, kh), s0);
      s0 = wmma_bf16(aq1, cvt_frag_f32(kp + 32, kh), s0);
      kp = Qh + (size_t)(j0 + 16 + lanelo) * 1024;
      s1 = wmma_bf16(aq0, cvt_frag_f32(kp, kh), s1);
      s1 = wmma_bf16(aq1, cvt_frag_f32(kp + 32, kh), s1);
    }

    // Online softmax per row; write P into LDS (row-major 16x32 bf16).
#pragma unroll
    for (int r = 0; r < 8; ++r) {
      float x0 = s0[r] * inv_scale;
      float x1 = s1[r] * inv_scale;
      float tm = rowmax16(fmaxf(x0, x1));
      float mn = fmaxf(m[r], tm);
      float p0 = __expf(x0 - mn);
      float p1 = __expf(x1 - mn);
      float f = __expf(m[r] - mn);
      float ps = rowsum16(p0 + p1);
      l[r] = l[r] * f + ps;
      m[r] = mn;
#pragma unroll
      for (int t = 0; t < 4; ++t) o[t][r] *= f;
      int prow = r + 8 * half;
      pbase[prow * 32 + lanelo] = (__bf16)p0;
      pbase[prow * 32 + 16 + lanelo] = (__bf16)p1;
    }

    // P as A-fragment (LDS bounce does the C->A relayout).
    v16bf pa = load_frag_bf16(pbase + lanelo * 32, kh);

    // V fragments from the transposed LDS tile: contiguous ds_load_b128 x2.
#pragma unroll
    for (int t = 0; t < 4; ++t) {
      v16bf vb = load_frag_bf16(vtile + (t * 16 + lanelo) * 32, kh);
      o[t] = wmma_bf16(pa, vb, o[t]);
    }
    __syncthreads();  // protect vtile before next j-step's stores
  }

  // Normalize and store (concat layout).
#pragma unroll
  for (int t = 0; t < 4; ++t) {
#pragma unroll
    for (int r = 0; r < 8; ++r) {
      int row = q0 + r + 8 * half;
      int col = h * 64 + t * 16 + lanelo;
      float v = o[t][r] / l[r];
      Zbf[((size_t)b * 1024 + row) * 1024 + col] = (__bf16)v;
    }
  }
}

// ---------------------------------------------------------------------------
// y = alpha * (x + r - mean) / (std_unbiased + 1e-6) + beta  (per row of 1024)
// ---------------------------------------------------------------------------
__global__ __launch_bounds__(256) void residual_ln_kernel(const float* __restrict__ X,
                                                          const float* __restrict__ R,
                                                          const float* __restrict__ alpha,
                                                          const float* __restrict__ beta,
                                                          float* outF, __bf16* outB) {
  __shared__ float s1[256], s2[256], stats[2];
  const int row = blockIdx.x;
  const int t = threadIdx.x;
  const float* xr = X + (size_t)row * 1024;
  const float* rr = R + (size_t)row * 1024;

  float v[4], sum = 0.0f, sq = 0.0f;
#pragma unroll
  for (int i = 0; i < 4; ++i) {
    int c = t + i * 256;
    float a = xr[c] + rr[c];
    v[i] = a; sum += a; sq += a * a;
  }
  s1[t] = sum; s2[t] = sq;
  __syncthreads();
  for (int s = 128; s > 0; s >>= 1) {
    if (t < s) { s1[t] += s1[t + s]; s2[t] += s2[t + s]; }
    __syncthreads();
  }
  if (t == 0) {
    float mean = s1[0] * (1.0f / 1024.0f);
    float var = (s2[0] - 1024.0f * mean * mean) * (1.0f / 1023.0f);  // ddof=1
    var = fmaxf(var, 0.0f);
    stats[0] = mean;
    stats[1] = 1.0f / (sqrtf(var) + 1e-6f);
  }
  __syncthreads();
  float mean = stats[0], rinv = stats[1];
#pragma unroll
  for (int i = 0; i < 4; ++i) {
    int c = t + i * 256;
    float y = alpha[c] * (v[i] - mean) * rinv + beta[c];
    size_t off = (size_t)row * 1024 + c;
    if (outF) outF[off] = y;
    if (outB) outB[off] = (__bf16)y;
  }
}

// ---------------------------------------------------------------------------
// Orchestration
// ---------------------------------------------------------------------------
extern "C" void kernel_launch(void* const* d_in, const int* in_sizes, int n_in,
                              void* d_out, int out_size, void* d_ws, size_t ws_size,
                              hipStream_t stream) {
  const float* x     = (const float*)d_in[0];   // [4,1024,1024]
  const float* Wq    = (const float*)d_in[1];   // [1024,1024]
  const float* bq    = (const float*)d_in[2];
  const float* Wo    = (const float*)d_in[3];
  const float* bo    = (const float*)d_in[4];
  const float* W1    = (const float*)d_in[5];   // [1024,2048]
  const float* b1    = (const float*)d_in[6];
  const float* W2    = (const float*)d_in[7];   // [2048,1024]
  const float* b2    = (const float*)d_in[8];
  const float* alpha = (const float*)d_in[9];
  const float* beta  = (const float*)d_in[10];

  char* ws = (char*)d_ws;
  const size_t MB = 1u << 20;
  __bf16* Xbf  = (__bf16*)(ws + 0);         //  8 MB  (4M bf16)
  __bf16* WqT  = (__bf16*)(ws + 8 * MB);    //  2 MB
  __bf16* WoT  = (__bf16*)(ws + 10 * MB);   //  2 MB
  __bf16* W1T  = (__bf16*)(ws + 12 * MB);   //  4 MB
  __bf16* W2T  = (__bf16*)(ws + 16 * MB);   //  4 MB
  float*  Qf   = (float*)(ws + 20 * MB);    // 16 MB
  __bf16* Zbf  = (__bf16*)(ws + 36 * MB);   //  8 MB
  float*  tmpF = (float*)(ws + 44 * MB);    // 16 MB (attn_out, then ff)
  float*  Yf   = (float*)(ws + 60 * MB);    // 16 MB
  __bf16* Ybf  = (__bf16*)(ws + 76 * MB);   //  8 MB
  __bf16* Hbf  = (__bf16*)(ws + 84 * MB);   // 16 MB (4096x2048)

  const int NTOK = 4096;                     // b*s

  // bf16 conversions / weight transposes
  cvt_kernel<<<(NTOK * 1024 + 255) / 256, 256, 0, stream>>>(x, Xbf, NTOK * 1024);
  transpose_cvt_kernel<<<(1024 * 1024 + 255) / 256, 256, 0, stream>>>(Wq, WqT, 1024, 1024);
  transpose_cvt_kernel<<<(1024 * 1024 + 255) / 256, 256, 0, stream>>>(Wo, WoT, 1024, 1024);
  transpose_cvt_kernel<<<(1024 * 2048 + 255) / 256, 256, 0, stream>>>(W1, W1T, 1024, 2048);
  transpose_cvt_kernel<<<(2048 * 1024 + 255) / 256, 256, 0, stream>>>(W2, W2T, 2048, 1024);

  // Q = X @ Wq + bq   (q = k = v in the reference)
  gemm_kernel<<<dim3(1024 / 64, NTOK / 128), 256, 0, stream>>>(
      Xbf, WqT, bq, Qf, nullptr, NTOK, 1024, 1024, 0);

  // Z = softmax(Q Q^T / 8) @ Q, concat layout, bf16
  attn_kernel<<<dim3(1024 / 64, 64), 128, 0, stream>>>(Qf, Zbf);

  // attn_out = Z @ Wo + bo
  gemm_kernel<<<dim3(1024 / 64, NTOK / 128), 256, 0, stream>>>(
      Zbf, WoT, bo, tmpF, nullptr, NTOK, 1024, 1024, 0);

  // Y = LN(x + attn_out)
  residual_ln_kernel<<<NTOK, 256, 0, stream>>>(x, tmpF, alpha, beta, Yf, Ybf);

  // H = relu(Y @ W1 + b1)
  gemm_kernel<<<dim3(2048 / 64, NTOK / 128), 256, 0, stream>>>(
      Ybf, W1T, b1, nullptr, Hbf, NTOK, 2048, 1024, 1);

  // ff = H @ W2 + b2
  gemm_kernel<<<dim3(16, NTOK / 128), 256, 0, stream>>>(
      Hbf, W2T, b2, tmpF, nullptr, NTOK, 1024, 2048, 0);

  // out = LN(Y + ff)
  residual_ln_kernel<<<NTOK, 256, 0, stream>>>(Yf, tmpF, alpha, beta, (float*)d_out, nullptr);
}
